// GAT_9036611190939
// MI455X (gfx1250) — compile-verified, matrix operand order checked by reading, not run on hardware
//
#include <hip/hip_runtime.h>
#include <hip/hip_bf16.h>

// ---------------- problem constants (match reference) ----------------
constexpr int N  = 50000;
constexpr int E  = 800000;
constexpr int D  = 256;
constexpr int H  = 8;
constexpr int ED = 16;
constexpr int L  = 6;
constexpr int D4 = 4 * D;       // 1024
constexpr float NEG_SLOPE = 0.2f;
constexpr float EPS = 1e-5f;

// ---------------- vector types ----------------
typedef __attribute__((ext_vector_type(16))) __bf16   v16bf;
typedef __attribute__((ext_vector_type(8)))  float    v8f;
typedef __attribute__((ext_vector_type(4)))  unsigned v4u;
typedef __attribute__((ext_vector_type(8)))  unsigned v8u;

// fp32 -> bf16 hi/lo split helpers (truncation; residual lands in lo term)
__device__ inline unsigned short bf_hi_bits(float f) {
    return (unsigned short)(__float_as_uint(f) >> 16);
}
__device__ inline float bf_bits_to_f32(unsigned short u) {
    return __uint_as_float(((unsigned)u) << 16);
}

// monotonic float<->uint encoding for atomicMax-based segment max
__device__ inline unsigned fenc(float f) {
    unsigned u = __float_as_uint(f);
    return (u & 0x80000000u) ? ~u : (u | 0x80000000u);
}
__device__ inline float fdec(unsigned u) {
    return __uint_as_float((u & 0x80000000u) ? (u & 0x7fffffffu) : ~u);
}

// =====================================================================
// Weight pre-pass: fp32 [K x Nc] row-major  ->  bf16 hi/lo chunks.
// Chunk = (128-col tile bn) x (32-k tile kc), stored column-major [c][k]:
//   chunk base (u16): ((bn*(K/32)+kc) * 8192)
//   hi at  c*32+k , lo at 4096 + c*32+k       (16 KB per chunk total)
// The LDS image (after TDM padding) is exactly the WMMA B-fragment order.
// =====================================================================
__global__ __launch_bounds__(256) void k_prep_w(
    const float* __restrict__ B, unsigned short* __restrict__ prep,
    int Nc, int K)
{
    int idx = blockIdx.x * 256 + threadIdx.x;
    if (idx >= K * Nc) return;
    int r = idx / Nc, col = idx - r * Nc;
    int bn = col >> 7, c = col & 127;
    int kc = r >> 5,  k = r & 31;
    size_t base = ((size_t)bn * (K >> 5) + kc) * 8192 + (size_t)c * 32 + k;
    float f = B[idx];
    unsigned short h = bf_hi_bits(f);
    prep[base] = h;
    prep[base + 4096] = bf_hi_bits(f - bf_bits_to_f32(h));
}

// =====================================================================
// Graph pre-pass kernels (a_edge is invariant across all 6 layers)
// =====================================================================
__global__ __launch_bounds__(256) void k_deg_easum(
    const long long* __restrict__ ei, const float* __restrict__ eattr,
    float* __restrict__ deg, float* __restrict__ easum)
{
    int e = blockIdx.x * 256 + threadIdx.x;
    if (e >= E) return;
    int d = (int)ei[E + e];
    atomicAdd(&deg[d], 1.0f);
    const float* ea = eattr + (size_t)e * ED;
    float* dst = easum + (size_t)d * ED;
#pragma unroll
    for (int k = 0; k < ED; k++) atomicAdd(&dst[k], ea[k]);
}

__global__ __launch_bounds__(256) void k_ea_mean(
    const float* __restrict__ deg, float* __restrict__ easum)
{
    int idx = blockIdx.x * 256 + threadIdx.x;
    if (idx >= N * ED) return;
    int n = idx >> 4;
    easum[idx] = easum[idx] / fmaxf(deg[n], 1.0f);
}

__global__ void k_vmat(const float* __restrict__ W_edge,
                       const float* __restrict__ att_edge,
                       float* __restrict__ vmat)
{
    int t = threadIdx.x;
    if (t >= ED * H) return;
    int k = t >> 3, h = t & 7;
    float s = 0.f;
    for (int c = 0; c < 32; c++) s += W_edge[k * D + h * 32 + c] * att_edge[h * 32 + c];
    vmat[k * H + h] = s;
}

__global__ __launch_bounds__(256) void k_a_edge(
    const float* __restrict__ eattr, const float* __restrict__ ea_mean,
    const float* __restrict__ vmat, float* __restrict__ a_edge)
{
    int e = blockIdx.x * 256 + threadIdx.x;
    if (e >= E + N) return;
    const float* ea = (e < E) ? (eattr + (size_t)e * ED)
                              : (ea_mean + (size_t)(e - E) * ED);
    float eav[ED];
#pragma unroll
    for (int k = 0; k < ED; k++) eav[k] = ea[k];
#pragma unroll
    for (int h = 0; h < H; h++) {
        float s = 0.f;
#pragma unroll
        for (int k = 0; k < ED; k++) s += eav[k] * vmat[k * H + h];
        a_edge[(size_t)e * H + h] = s;
    }
}

// =====================================================================
// WMMA GEMM: C[M x NC] = A[M x K] @ B[K x NC], fp32 A/C, prepped bf16 B.
// Block = 256 threads (8 wave32s): 64-row strip x 128-col tile.
// B chunks (16 KB hi+lo) are DMA'd into double-buffered LDS by the TDM
// (pad_enable: +16 B every 64 B -> 80 B row stride, bank-stride 20);
// chunk kc+1 is issued before waiting on chunk kc (s_wait_tensorcnt 1,
// TDM completes in-order per wave). Per wave per chunk: 4 M-subtiles x
// 3x v_wmma_f32_16x16x32_bf16 (bf16x3 fp32 emulation). Row clamping is
// confined to staging/epilogue so every WMMA runs with EXEC all-ones.
// =====================================================================
#define SB_ROW 40        /* u16 per padded 32-k row (64 B data + 16 B pad) */
#define SB_LO  5120      /* u16 offset of lo half: 128 rows * 80 B / 2     */

__device__ inline void issue_tdm(unsigned long long ga, unsigned sb_lds)
{
    v4u g0;
    g0[0] = 1u;                                   // count=1 (valid user D#)
    g0[1] = sb_lds;                               // lds_addr
    g0[2] = (unsigned)(ga & 0xffffffffull);       // global_addr[31:0]
    g0[3] = (unsigned)((ga >> 32) & 0x1ffffffull) // global_addr[56:32]
          | 0x80000000u;                          // type=2 ("image")
    v8u g1;
    g1[0] = (3u << 16)      // data_size=3 (8-byte elements)
          | (1u << 20)      // pad_enable
          | (3u << 22)      // pad_interval: 2^(3+1)=16 dwords (64 B)
          | (3u << 25);     // pad_amount: 3+1 = 4 dwords (16 B)
    g1[1] = 2048u << 16;    // tensor_dim0 = 2048 elements (16 KB)
    g1[2] = 1u << 16;       // tensor_dim1 = 1
    g1[3] = 2048u << 16;    // tile_dim0 = 2048
    g1[4] = 1u;             // tile_dim1 = 1
    g1[5] = 2048u;          // tensor_dim0_stride
    g1[6] = 0u;
    g1[7] = 0u;
    asm volatile("tensor_load_to_lds %0, %1" :: "s"(g0), "s"(g1) : "memory");
}

template<int NC, int K>
__global__ __launch_bounds__(256) void k_gemm_bf16x3(
    const float* __restrict__ A, const unsigned short* __restrict__ Bprep,
    float* __restrict__ Cout, const float* __restrict__ bias, int do_relu)
{
    constexpr int NKC = K / 32;
    __shared__ unsigned short sAhi[64 * 32], sAlo[64 * 32];  // 4 KB + 4 KB
    __shared__ unsigned short sB[2][2 * SB_LO];              // 2 x 20 KB

    const int tid  = threadIdx.x;
    const int bm   = blockIdx.x;      // 64-row strip
    const int bn   = blockIdx.y;      // 128-col tile
    const int lane = tid & 31;
    const int wv   = tid >> 5;        // wave -> 16-col sub-tile

    const unsigned long long bbase =
        (unsigned long long)Bprep + (unsigned long long)bn * NKC * 16384ull;

    v8f acc[4] = {};

    if (wv == 0)
        issue_tdm(bbase, (unsigned)(unsigned long long)(const void*)&sB[0][0]);

    for (int kc = 0; kc < NKC; kc++) {
        // prefetch next B chunk into the other buffer while we stage A
        if (wv == 0 && kc + 1 < NKC)
            issue_tdm(bbase + (unsigned long long)(kc + 1) * 16384ull,
                      (unsigned)(unsigned long long)(const void*)&sB[(kc + 1) & 1][0]);

        // stage A 64x32 (fp32 -> bf16 hi/lo); rows clamped at N-1.
        // One float4 (global_load_b128) + two 8-byte LDS stores per half.
#pragma unroll
        for (int half = 0; half < 2; half++) {
            int r   = (tid >> 3) + half * 32;     // 0..63
            int c4  = (tid & 7) * 4;              // 0,4,...,28
            int row = bm * 64 + r;
            if (row > N - 1) row = N - 1;
            const float4 f4 = *(const float4*)(A + (size_t)row * K + kc * 32 + c4);
            unsigned short h0 = bf_hi_bits(f4.x), h1 = bf_hi_bits(f4.y);
            unsigned short h2 = bf_hi_bits(f4.z), h3 = bf_hi_bits(f4.w);
            ushort4 hv, lv;
            hv.x = h0; hv.y = h1; hv.z = h2; hv.w = h3;
            lv.x = bf_hi_bits(f4.x - bf_bits_to_f32(h0));
            lv.y = bf_hi_bits(f4.y - bf_bits_to_f32(h1));
            lv.z = bf_hi_bits(f4.z - bf_bits_to_f32(h2));
            lv.w = bf_hi_bits(f4.w - bf_bits_to_f32(h3));
            *(ushort4*)&sAhi[r * 32 + c4] = hv;
            *(ushort4*)&sAlo[r * 32 + c4] = lv;
        }
        if (wv == 0) {
            if (kc + 1 < NKC) __builtin_amdgcn_s_wait_tensorcnt(1);
            else              __builtin_amdgcn_s_wait_tensorcnt(0);
        }
        __syncthreads();

        // ---- fragments: contiguous 16-byte LDS loads ----
        // lane l, slot j -> K = ((j<8)? j : j+8) + ((l>=16)? 8 : 0)
        const unsigned short* sb = sB[kc & 1];
        const int koff = (lane >> 4) << 3;
        const int bc   = wv * 16 + (lane & 15);

        union U { v16bf v; uint4 q[2]; };
        U bhi, blo;
        const unsigned short* pb  = sb + bc * SB_ROW;
        const unsigned short* pbl = sb + SB_LO + bc * SB_ROW;
        bhi.q[0] = *(const uint4*)(pb  + koff);
        bhi.q[1] = *(const uint4*)(pb  + koff + 16);
        blo.q[0] = *(const uint4*)(pbl + koff);
        blo.q[1] = *(const uint4*)(pbl + koff + 16);

#pragma unroll
        for (int sub = 0; sub < 4; sub++) {
            U ahi, alo;
            const unsigned short* pa  = &sAhi[(sub * 16 + (lane & 15)) * 32];
            const unsigned short* pal = &sAlo[(sub * 16 + (lane & 15)) * 32];
            ahi.q[0] = *(const uint4*)(pa  + koff);
            ahi.q[1] = *(const uint4*)(pa  + koff + 16);
            alo.q[0] = *(const uint4*)(pal + koff);
            alo.q[1] = *(const uint4*)(pal + koff + 16);

            // bf16x3 emulation of fp32: Ahi*Bhi + Ahi*Blo + Alo*Bhi
            acc[sub] = __builtin_amdgcn_wmma_f32_16x16x32_bf16(
                false, ahi.v, false, bhi.v, (short)0, acc[sub], false, false);
            acc[sub] = __builtin_amdgcn_wmma_f32_16x16x32_bf16(
                false, ahi.v, false, blo.v, (short)0, acc[sub], false, false);
            acc[sub] = __builtin_amdgcn_wmma_f32_16x16x32_bf16(
                false, alo.v, false, bhi.v, (short)0, acc[sub], false, false);
        }
        __syncthreads();
    }

    // epilogue: VGPR r, lane l -> row = r + (l>=16 ? 8:0), col = l&15
    const int col = bn * 128 + wv * 16 + (lane & 15);
    float bv = bias ? bias[col] : 0.f;
#pragma unroll
    for (int sub = 0; sub < 4; sub++) {
        int rbase = bm * 64 + sub * 16 + ((lane >> 4) << 3);
#pragma unroll
        for (int r = 0; r < 8; r++) {
            int row = rbase + r;
            if (row < N) {
                float v = acc[sub][r] + bv;
                if (do_relu) v = fmaxf(v, 0.f);
                Cout[(size_t)row * NC + col] = v;
            }
        }
    }
}

// =====================================================================
// Per-layer attention kernels
// =====================================================================
__global__ __launch_bounds__(256) void k_a_srcdst(
    const float* __restrict__ xp, const float* __restrict__ att_src,
    const float* __restrict__ att_dst, float* __restrict__ a_src,
    float* __restrict__ a_dst)
{
    int idx = blockIdx.x * 256 + threadIdx.x;
    if (idx >= N * H) return;
    int n = idx >> 3, h = idx & 7;
    const float* r = xp + (size_t)n * D + h * 32;
    float s1 = 0.f, s2 = 0.f;
#pragma unroll
    for (int c = 0; c < 32; c++) {
        float v = r[c];
        s1 += v * att_src[h * 32 + c];
        s2 += v * att_dst[h * 32 + c];
    }
    a_src[idx] = s1;
    a_dst[idx] = s2;
}

__global__ __launch_bounds__(256) void k_alpha_max(
    const long long* __restrict__ ei, const float* __restrict__ a_src,
    const float* __restrict__ a_dst, const float* __restrict__ a_edge,
    float* __restrict__ alpha, unsigned* __restrict__ amax)
{
    int e = blockIdx.x * 256 + threadIdx.x;
    if (e >= E + N) return;
    int s, d;
    if (e < E) { s = (int)ei[e]; d = (int)ei[E + e]; }
    else       { s = d = e - E; }
#pragma unroll
    for (int h = 0; h < H; h++) {
        float al = a_src[s * H + h] + a_dst[d * H + h] + a_edge[(size_t)e * H + h];
        al = (al > 0.f) ? al : NEG_SLOPE * al;     // leaky relu
        alpha[(size_t)e * H + h] = al;
        atomicMax(&amax[d * H + h], fenc(al));     // segment max
    }
}

__global__ __launch_bounds__(256) void k_exp_denom(
    const long long* __restrict__ ei, float* __restrict__ alpha,
    const unsigned* __restrict__ amax, float* __restrict__ denom)
{
    int idx = blockIdx.x * 256 + threadIdx.x;
    if (idx >= (E + N) * H) return;
    int e = idx >> 3, h = idx & 7;
    int d = (e < E) ? (int)ei[E + e] : (e - E);
    float ex = __expf(alpha[idx] - fdec(amax[d * H + h]));
    alpha[idx] = ex;                               // overwrite with exp
    atomicAdd(&denom[d * H + h], ex);
}

// message pass: 64 threads/edge, 4 channels each (float4 gather, 4 atomics)
__global__ __launch_bounds__(256) void k_scatter_msg(
    const long long* __restrict__ ei, const float* __restrict__ ex,
    const float* __restrict__ denom, const float* __restrict__ xp,
    float* __restrict__ outp)
{
    int eg = blockIdx.x * 4 + (threadIdx.x >> 6);
    if (eg >= E + N) return;
    int t = threadIdx.x & 63;
    int s, d;
    if (eg < E) { s = (int)ei[eg]; d = (int)ei[E + eg]; }
    else        { s = d = eg - E; }
    int h  = t >> 3;            // 4 channels per thread stay within one head
    int c0 = t * 4;
    float attn = ex[(size_t)eg * H + h] / denom[d * H + h];
    const float4 xv = *(const float4*)(xp + (size_t)s * D + c0);
    float* po = outp + (size_t)d * D + c0;
    atomicAdd(po + 0, xv.x * attn);
    atomicAdd(po + 1, xv.y * attn);
    atomicAdd(po + 2, xv.z * attn);
    atomicAdd(po + 3, xv.w * attn);
}

// =====================================================================
// Fused (xa + xb + colbias) -> LayerNorm(g,b). One wave32 per 256-wide row.
// =====================================================================
__global__ __launch_bounds__(256) void k_ln(
    const float* __restrict__ xa, const float* __restrict__ xb,
    const float* __restrict__ colb, const float* __restrict__ g,
    const float* __restrict__ b, float* __restrict__ outp)
{
    int lane = threadIdx.x & 31;
    int row  = blockIdx.x * 8 + (threadIdx.x >> 5);   // N divisible by 8
    const float* pa = xa + (size_t)row * D;
    const float* pb = xb + (size_t)row * D;
    float t[8];
    float s = 0.f, ss = 0.f;
#pragma unroll
    for (int k = 0; k < 8; k++) {
        int c = lane + k * 32;
        float v = pa[c] + pb[c] + (colb ? colb[c] : 0.f);
        t[k] = v; s += v; ss += v * v;
    }
#pragma unroll
    for (int m = 16; m >= 1; m >>= 1) {
        s  += __shfl_xor(s,  m, 32);
        ss += __shfl_xor(ss, m, 32);
    }
    float mu  = s * (1.f / D);
    float var = ss * (1.f / D) - mu * mu;
    float rs  = rsqrtf(var + EPS);
    float* po = outp + (size_t)row * D;
#pragma unroll
    for (int k = 0; k < 8; k++) {
        int c = lane + k * 32;
        po[c] = (t[k] - mu) * rs * g[c] + b[c];
    }
}

// =====================================================================
// Host driver
// =====================================================================
extern "C" void kernel_launch(void* const* d_in, const int* in_sizes, int n_in,
                              void* d_out, int out_size, void* d_ws, size_t ws_size,
                              hipStream_t stream)
{
    (void)in_sizes; (void)n_in; (void)out_size; (void)ws_size;

    const float*     x        = (const float*)d_in[0];
    const long long* ei       = (const long long*)d_in[1];
    const float*     eattr    = (const float*)d_in[2];
    const float*     W        = (const float*)d_in[3];
    const float*     att_src  = (const float*)d_in[4];
    const float*     att_dst  = (const float*)d_in[5];
    const float*     att_edge = (const float*)d_in[6];
    const float*     W_edge   = (const float*)d_in[7];
    const float*     bias     = (const float*)d_in[8];
    const float*     ffn_w1   = (const float*)d_in[9];
    const float*     ffn_b1   = (const float*)d_in[10];
    const float*     ffn_w2   = (const float*)d_in[11];
    const float*     ffn_w2b  = (const float*)d_in[12];   // ffn_b2
    const float*     ln1_g    = (const float*)d_in[13];
    const float*     ln1_b    = (const float*)d_in[14];
    const float*     ln2_g    = (const float*)d_in[15];
    const float*     ln2_b    = (const float*)d_in[16];

    // workspace carve-up (~437 MB)
    float* p = (float*)d_ws;
    size_t o = 0;
    float*    buf_m   = p + o; o += (size_t)N * D;
    float*    x_proj  = p + o; o += (size_t)N * D;
    float*    gbuf    = p + o; o += (size_t)N * D;        // GAT out / FFN h2
    float*    hidden  = p + o; o += (size_t)N * D4;
    float*    a_src   = p + o; o += (size_t)N * H;
    float*    a_dst   = p + o; o += (size_t)N * H;
    float*    denom   = p + o; o += (size_t)N * H;
    unsigned* amax    = (unsigned*)(p + o); o += (size_t)N * H;
    float*    deg     = p + o; o += (size_t)N;
    float*    ea_mean = p + o; o += (size_t)N * ED;       // easum then mean
    float*    vmat    = p + o; o += (size_t)ED * H;
    float*    a_edge  = p + o; o += (size_t)(E + N) * H;
    float*    alpha   = p + o; o += (size_t)(E + N) * H;  // alpha then exp
    // prepped bf16 hi/lo weights (u16), ~12.9 MB
    unsigned short* prepW  = (unsigned short*)(p + o);
    unsigned short* prep1  = prepW + (size_t)2 * D * D;
    unsigned short* prep2  = prep1 + (size_t)2 * L * D * D4;

    const int GM = (N + 63) / 64;   // 64-row GEMM strips (782)

    // m = x
    hipMemcpyAsync(buf_m, x, (size_t)N * D * sizeof(float),
                   hipMemcpyDeviceToDevice, stream);

    // ---- one-time weight prep (fp32 -> chunked bf16 hi/lo) ----
    k_prep_w<<<(D * D + 255) / 256, 256, 0, stream>>>(W, prepW, D, D);
    for (int i = 0; i < L; i++) {
        k_prep_w<<<(D * D4 + 255) / 256, 256, 0, stream>>>(
            ffn_w1 + (size_t)i * D * D4, prep1 + (size_t)2 * i * D * D4, D4, D);
        k_prep_w<<<(D4 * D + 255) / 256, 256, 0, stream>>>(
            ffn_w2 + (size_t)i * D4 * D, prep2 + (size_t)2 * i * D4 * D, D, D4);
    }

    // ---- one-time graph pre-pass (layer-invariant attention edge term) ----
    hipMemsetAsync(deg, 0, (size_t)N * sizeof(float), stream);
    hipMemsetAsync(ea_mean, 0, (size_t)N * ED * sizeof(float), stream);
    k_deg_easum<<<(E + 255) / 256, 256, 0, stream>>>(ei, eattr, deg, ea_mean);
    k_ea_mean<<<(N * ED + 255) / 256, 256, 0, stream>>>(deg, ea_mean);
    k_vmat<<<1, 128, 0, stream>>>(W_edge, att_edge, vmat);
    k_a_edge<<<(E + N + 255) / 256, 256, 0, stream>>>(eattr, ea_mean, vmat, a_edge);

    for (int i = 0; i < L; i++) {
        // x_proj = m @ W  (WMMA + double-buffered TDM)
        k_gemm_bf16x3<D, D><<<dim3(GM, D / 128), 256, 0, stream>>>(
            buf_m, prepW, x_proj, nullptr, 0);

        k_a_srcdst<<<(N * H + 255) / 256, 256, 0, stream>>>(
            x_proj, att_src, att_dst, a_src, a_dst);

        hipMemsetAsync(amax,  0, (size_t)N * H * sizeof(unsigned), stream);
        hipMemsetAsync(denom, 0, (size_t)N * H * sizeof(float), stream);
        hipMemsetAsync(gbuf,  0, (size_t)N * D * sizeof(float), stream);

        k_alpha_max<<<(E + N + 255) / 256, 256, 0, stream>>>(
            ei, a_src, a_dst, a_edge, alpha, amax);
        k_exp_denom<<<((E + N) * H + 255) / 256, 256, 0, stream>>>(
            ei, alpha, amax, denom);
        k_scatter_msg<<<(E + N + 3) / 4, 256, 0, stream>>>(
            ei, alpha, denom, x_proj, gbuf);

        // m = LN1(gat_out + bias + m)
        k_ln<<<N / 8, 256, 0, stream>>>(gbuf, buf_m, bias,
                                        ln1_g + (size_t)i * D,
                                        ln1_b + (size_t)i * D, buf_m);

        // hidden = relu(m @ W1 + b1)   (WMMA + TDM, fused bias+relu)
        k_gemm_bf16x3<D4, D><<<dim3(GM, D4 / 128), 256, 0, stream>>>(
            buf_m, prep1 + (size_t)2 * i * D * D4, hidden,
            ffn_b1 + (size_t)i * D4, 1);

        // h2 = hidden @ W2   (WMMA + TDM)
        k_gemm_bf16x3<D, D4><<<dim3(GM, D / 128), 256, 0, stream>>>(
            hidden, prep2 + (size_t)2 * i * D4 * D, gbuf, nullptr, 0);

        // m = LN2(h2 + b2 + m); last layer writes d_out
        float* mout = (i == L - 1) ? (float*)d_out : buf_m;
        k_ln<<<N / 8, 256, 0, stream>>>(gbuf, buf_m,
                                        ffn_w2b + (size_t)i * D,
                                        ln2_g + (size_t)i * D,
                                        ln2_b + (size_t)i * D, mout);
    }
}
